// MultiHeadAttention_83116207112396
// MI455X (gfx1250) — matrix operand order repaired and checked by reading, not compiled, not run on hardware
//
#include <hip/hip_runtime.h>

// ---------------------------------------------------------------------------
// MultiHeadAttention forward for MI455X (gfx1250, wave32, WMMA).
// B=32, L=512, D_IN=D_ATT=512, H=8, depth=64.
// Pass 0: masks + weight transpose/convert to bf16.
// Pass 1: Q/K/V projection GEMMs via v_wmma_f32_16x16x32_bf16; each wave owns
//         a 16x64 tile (A-fragment reused across 4 accumulators).
// Pass 2: flash-attention streaming pass; block cooperatively stages K/V
//         tiles into LDS with GLOBAL_LOAD_ASYNC_TO_LDS_B128 (ASYNCcnt),
//         online softmax in wave32 half-lane reductions, P transposed
//         C-layout -> A-layout through LDS, WMMA for QK^T and PV.
// ---------------------------------------------------------------------------

typedef __attribute__((ext_vector_type(16))) __bf16 bf16x16;
typedef __attribute__((ext_vector_type(8)))  __bf16 bf16x8;
typedef __attribute__((ext_vector_type(8)))  float  v8f;
typedef int v4i __attribute__((vector_size(16)));   // matches builtin's V4i

#define BATCH 32
#define SEQ   512
#define DIN   512
#define HEADS 8
#define DEPTH 64
#define NEGF  (-4294967295.0f)   /* -2^32 + 1, reference mask fill */

#if __has_builtin(__builtin_amdgcn_global_load_async_to_lds_b128)
#define USE_ASYNC_LDS 1
#else
#define USE_ASYNC_LDS 0
#endif

__device__ __forceinline__ v8f zero8() {
  v8f z = {0.f, 0.f, 0.f, 0.f, 0.f, 0.f, 0.f, 0.f};
  return z;
}

// A-fragment (16x32 bf16): lane l holds row m=l%16; elements 0..7 = K k0..k0+7,
// elements 8..15 = K k0+16..k0+23, with k0 = 8*(l/16).
__device__ __forceinline__ bf16x16 load_a_bf16(const __bf16* base, int ld,
                                               int mbase, int kbase, int lane) {
  int m  = lane & 15;
  int k0 = kbase + ((lane >> 4) << 3);
  const __bf16* p = base + (size_t)(mbase + m) * ld + k0;
  bf16x8 lo = *(const bf16x8*)(p);
  bf16x8 hi = *(const bf16x8*)(p + 16);
  return __builtin_shufflevector(lo, hi, 0,1,2,3,4,5,6,7,8,9,10,11,12,13,14,15);
}

// Same A-fragment but converting from f32 on the fly (native bf16 converts).
__device__ __forceinline__ bf16x16 load_a_f32(const float* __restrict__ base,
                                              int ld, int mbase, int kbase, int lane) {
  int m  = lane & 15;
  int k0 = kbase + ((lane >> 4) << 3);
  const float* p = base + (size_t)(mbase + m) * ld + k0;
  bf16x16 a;
#pragma unroll
  for (int j = 0; j < 8; ++j) {
    a[j]     = (__bf16)p[j];
    a[8 + j] = (__bf16)p[16 + j];
  }
  return a;
}

// B-fragment (32x16 bf16) from a "B-transposed" row-major array bt[N][Kdim]
// (works for global or LDS pointers): lane l covers N = nbase + l%16,
// elements 0..7 = Kdim kbase + 8*(l/16) + j, elements 8..15 = +16.
__device__ __forceinline__ bf16x16 load_b_frag(const __bf16* bt, int ld,
                                               int nbase, int kbase, int lane) {
  int n  = nbase + (lane & 15);
  int kk = kbase + ((lane >> 4) << 3);
  const __bf16* p = bt + (size_t)n * ld + kk;
  bf16x8 lo = *(const bf16x8*)(p);
  bf16x8 hi = *(const bf16x8*)(p + 16);
  return __builtin_shufflevector(lo, hi, 0,1,2,3,4,5,6,7,8,9,10,11,12,13,14,15);
}

__device__ __forceinline__ float half16_max(float v) {
  v = fmaxf(v, __shfl_xor(v, 1, 32));
  v = fmaxf(v, __shfl_xor(v, 2, 32));
  v = fmaxf(v, __shfl_xor(v, 4, 32));
  v = fmaxf(v, __shfl_xor(v, 8, 32));
  return v;
}
__device__ __forceinline__ float half16_sum(float v) {
  v += __shfl_xor(v, 1, 32);
  v += __shfl_xor(v, 2, 32);
  v += __shfl_xor(v, 4, 32);
  v += __shfl_xor(v, 8, 32);
  return v;
}

// 16-byte async global->LDS copy (per-lane addresses), with sync fallback.
__device__ __forceinline__ void cp16_g2s(const __bf16* g, __bf16* l) {
#if USE_ASYNC_LDS
  __builtin_amdgcn_global_load_async_to_lds_b128(
      (__attribute__((address_space(1))) v4i*)g,
      (__attribute__((address_space(3))) v4i*)l, 0, 0);
#else
  *(bf16x8*)l = *(const bf16x8*)g;
#endif
}
__device__ __forceinline__ void wait_async_cp() {
#if USE_ASYNC_LDS
#if __has_builtin(__builtin_amdgcn_s_wait_asynccnt)
  __builtin_amdgcn_s_wait_asynccnt(0);
#else
  asm volatile("s_wait_asynccnt 0" ::: "memory");
#endif
#endif
}

// ---------------------------------------------------------------------------
// Kernel 0a: padding masks. qmask[b*L+l] = (sum_d query != 0), same for key.
// ---------------------------------------------------------------------------
__global__ void mha_mask_kernel(const float* __restrict__ q,
                                const float* __restrict__ k,
                                float* __restrict__ qmask,
                                float* __restrict__ kmask) {
  int lane = threadIdx.x & 31;
  int wave = threadIdx.x >> 5;
  int row  = blockIdx.x * 8 + wave;          // 0 .. B*L-1
  const float* qp = q + (size_t)row * DIN;
  const float* kp = k + (size_t)row * DIN;
  float sq = 0.f, sk = 0.f;
#pragma unroll
  for (int i = lane; i < DIN; i += 32) { sq += qp[i]; sk += kp[i]; }
#pragma unroll
  for (int off = 16; off > 0; off >>= 1) {
    sq += __shfl_xor(sq, off, 32);
    sk += __shfl_xor(sk, off, 32);
  }
  if (lane == 0) {
    qmask[row] = (sq != 0.f) ? 1.f : 0.f;
    kmask[row] = (sk != 0.f) ? 1.f : 0.f;
  }
}

// ---------------------------------------------------------------------------
// Kernel 0b: Wt[which][n][k] = bf16(W[k][n])  (transpose + convert, 512x512 x3)
// ---------------------------------------------------------------------------
__global__ void mha_wt_kernel(const float* __restrict__ Wq,
                              const float* __restrict__ Wk,
                              const float* __restrict__ Wv,
                              __bf16* __restrict__ Wt) {
  int idx   = blockIdx.x * 256 + threadIdx.x;    // over 512*512
  int which = blockIdx.y;
  const float* W = (which == 0) ? Wq : (which == 1) ? Wk : Wv;
  int n = idx >> 9;
  int k = idx & 511;
  Wt[(size_t)which * (DIN * DIN) + idx] = (__bf16)W[(size_t)k * DIN + n];
}

// ---------------------------------------------------------------------------
// Kernel 1: projection GEMMs.  M = B*L = 16384, N = 512, K = 512.
// Each wave computes a 16x64 output tile: the converted A-fragment is reused
// across 4 WMMA accumulators (4x less f32->bf16 conversion + A traffic).
// which 0: Qh[h][b][l][d];  1: Kh[h][b][l][d];  2: Vt[h][b][d][l] (transposed)
// ---------------------------------------------------------------------------
__global__ void mha_proj_kernel(const float* __restrict__ query,
                                const float* __restrict__ key,
                                const __bf16* __restrict__ Wt,
                                __bf16* __restrict__ Qh,
                                __bf16* __restrict__ Kh,
                                __bf16* __restrict__ Vt) {
  int lane  = threadIdx.x & 31;
  int wave  = threadIdx.x >> 5;
  int mtile = blockIdx.x * 4 + wave;             // 0 .. 1023
  int nblk  = blockIdx.y * 64;                   // 0 .. 448
  int which = blockIdx.z;                        // 0,1,2
  int mbase = mtile * 16;

  const float*  A  = (which == 0) ? query : key;
  const __bf16* Bt = Wt + (size_t)which * (DIN * DIN);

  v8f acc[4] = {zero8(), zero8(), zero8(), zero8()};
#pragma unroll 4
  for (int kb = 0; kb < DIN; kb += 32) {
    bf16x16 af = load_a_f32(A, DIN, mbase, kb, lane);
#pragma unroll
    for (int c = 0; c < 4; ++c) {
      bf16x16 bf = load_b_frag(Bt, DIN, nblk + c * 16, kb, lane);
      acc[c] = __builtin_amdgcn_wmma_f32_16x16x32_bf16(false, af, false, bf,
                                                       (short)0, acc[c], false, false);
    }
  }

  // C layout: VGPR r -> M = r + 8*(lane/16), N = lane%16.
#pragma unroll
  for (int c = 0; c < 4; ++c) {
    int n = nblk + c * 16 + (lane & 15);
    int h = n >> 6;
    int d = n & 63;
#pragma unroll
    for (int r = 0; r < 8; ++r) {
      int row = mbase + r + ((lane >> 4) << 3);
      int b   = row >> 9;          // / SEQ
      int l   = row & 511;         // % SEQ
      __bf16 v = (__bf16)acc[c][r];
      size_t hb = (size_t)(h * BATCH + b);
      if (which == 0)      Qh[((hb * SEQ + l) << 6) + d] = v;
      else if (which == 1) Kh[((hb * SEQ + l) << 6) + d] = v;
      else                 Vt[((hb * DEPTH + d) << 9) + l] = v;
    }
  }
}

// ---------------------------------------------------------------------------
// Kernel 2: streaming attention + head-merge + residual.
// Block = 4 waves = 64 query rows of one (h,b); per 32-key step the block
// cooperatively stages the K tile (32x64 bf16, contiguous 4KB) and V tile
// (64x32 bf16) into LDS via async global->LDS copies, then each wave does
// 4 WMMA (scores) + online softmax + P via LDS + 4 WMMA (PV).
// Waves whose causal range has ended skip compute (wave-uniform branch) but
// keep staging + barriers.
// ---------------------------------------------------------------------------
__global__ void mha_attn_kernel(const __bf16* __restrict__ Qh,
                                const __bf16* __restrict__ Kh,
                                const __bf16* __restrict__ Vt,
                                const float* __restrict__ qmask,
                                const float* __restrict__ kmask,
                                const float* __restrict__ query,
                                float* __restrict__ out) {
  __shared__ __align__(16) __bf16 Kt[32 * 64];       // [key][depth], 4 KB
  __shared__ __align__(16) __bf16 Vs[64 * 32];       // [depth][key], 4 KB
  __shared__ __align__(16) __bf16 Pl[4][16][32];     // per-wave P tiles, 4 KB

  const int tid   = threadIdx.x;
  const int lane  = tid & 31;
  const int wave  = tid >> 5;
  const int hb    = blockIdx.x;                      // h*BATCH + b
  const int h     = hb >> 5;
  const int b     = hb & 31;
  const int qblk  = blockIdx.y * 64;
  const int qbase = qblk + wave * 16;

  const __bf16* Qp = Qh + (size_t)hb * SEQ * DEPTH;
  const __bf16* Kp = Kh + (size_t)hb * SEQ * DEPTH;
  const __bf16* Vp = Vt + (size_t)hb * DEPTH * SEQ;
  const float*  km = kmask + (size_t)b * SEQ;

  // Q fragments held in registers across the whole key sweep (depth 64 = 2).
  bf16x16 qa0 = load_a_bf16(Qp, DEPTH, qbase, 0,  lane);
  bf16x16 qa1 = load_a_bf16(Qp, DEPTH, qbase, 32, lane);

  float row_max[8], row_sum[8];
#pragma unroll
  for (int r = 0; r < 8; ++r) { row_max[r] = -__builtin_inff(); row_sum[r] = 0.f; }
  v8f o0 = zero8(), o1 = zero8(), o2 = zero8(), o3 = zero8();

  const float scale = 0.125f;                        // 1/sqrt(64)
  const int kend = qblk + 64;                        // block-uniform causal bound

  // Staging addresses (fixed per thread): K tile is one contiguous 4KB run;
  // V tile is 64 rows x 64B out of depth-major Vt rows (stride SEQ).
  const int vr = tid >> 2;                           // 0..31
  const int vc = (tid & 3) * 8;                      // 0,8,16,24 (elements)

  for (int kb = 0; kb < kend; kb += 32) {
    if (kb + 32 < kend)
      __builtin_prefetch(Kp + (size_t)(kb + 32) * DEPTH, 0, 1);

    // ---- cooperative stage of K and V tiles into LDS -------------------
    const __bf16* gk = Kp + (size_t)kb * DEPTH;      // 2048 contiguous elems
    cp16_g2s(gk + tid * 8,        Kt + tid * 8);
    cp16_g2s(gk + 1024 + tid * 8, Kt + 1024 + tid * 8);
    cp16_g2s(Vp + (size_t)vr * SEQ        + kb + vc, Vs + vr * 32 + vc);
    cp16_g2s(Vp + (size_t)(vr + 32) * SEQ + kb + vc, Vs + (vr + 32) * 32 + vc);
    wait_async_cp();
    __syncthreads();

    if (kb < qbase + 16) {                           // wave-uniform: still in range
      // ---- scores S = Q K^T for keys [kb, kb+32) -----------------------
      v8f s0 = zero8(), s1 = zero8();
      {
        bf16x16 k00 = load_b_frag(Kt, DEPTH, 0,  0,  lane);
        bf16x16 k01 = load_b_frag(Kt, DEPTH, 0,  32, lane);
        s0 = __builtin_amdgcn_wmma_f32_16x16x32_bf16(false, qa0, false, k00,
                                                     (short)0, s0, false, false);
        s0 = __builtin_amdgcn_wmma_f32_16x16x32_bf16(false, qa1, false, k01,
                                                     (short)0, s0, false, false);
        bf16x16 k10 = load_b_frag(Kt, DEPTH, 16, 0,  lane);
        bf16x16 k11 = load_b_frag(Kt, DEPTH, 16, 32, lane);
        s1 = __builtin_amdgcn_wmma_f32_16x16x32_bf16(false, qa0, false, k10,
                                                     (short)0, s1, false, false);
        s1 = __builtin_amdgcn_wmma_f32_16x16x32_bf16(false, qa1, false, k11,
                                                     (short)0, s1, false, false);
      }

      // ---- masks + online softmax (C layout) ---------------------------
      int   col0 = kb + (lane & 15);
      int   col1 = col0 + 16;
      float km0  = km[col0];
      float km1  = km[col1];
      int   mrow = ((lane >> 4) << 3);               // +8 for upper half-lanes

#pragma unroll
      for (int r = 0; r < 8; ++r) {
        int   row = qbase + r + mrow;
        float v0  = s0[r] * scale;
        float v1  = s1[r] * scale;
        if (col0 > row || km0 == 0.f) v0 = NEGF;
        if (col1 > row || km1 == 0.f) v1 = NEGF;

        float mloc = half16_max(fmaxf(v0, v1));
        float mnew = fmaxf(row_max[r], mloc);
        float corr = __expf(row_max[r] - mnew);
        float p0   = __expf(v0 - mnew);
        float p1   = __expf(v1 - mnew);
        row_sum[r] = row_sum[r] * corr + half16_sum(p0 + p1);
        row_max[r] = mnew;
        o0[r] *= corr; o1[r] *= corr; o2[r] *= corr; o3[r] *= corr;

        Pl[wave][r + mrow][(lane & 15)]      = (__bf16)p0;
        Pl[wave][r + mrow][(lane & 15) + 16] = (__bf16)p1;
      }

      // wave-private LDS tile: DS in-order per wave, just drain the counter
      asm volatile("s_wait_dscnt 0" ::: "memory");

      // ---- re-read P in A layout (16x32) -------------------------------
      bf16x16 pa;
      {
        int m  = lane & 15;
        int k0 = ((lane >> 4) << 3);
        bf16x8 lo = *(const bf16x8*)(&Pl[wave][m][k0]);
        bf16x8 hi = *(const bf16x8*)(&Pl[wave][m][k0 + 16]);
        pa = __builtin_shufflevector(lo, hi, 0,1,2,3,4,5,6,7,8,9,10,11,12,13,14,15);
      }

      // ---- O += P V  (Vs depth-major: rows = depth, cols = keys) -------
      bf16x16 vb0 = load_b_frag(Vs, 32, 0,  0, lane);
      o0 = __builtin_amdgcn_wmma_f32_16x16x32_bf16(false, pa, false, vb0,
                                                   (short)0, o0, false, false);
      bf16x16 vb1 = load_b_frag(Vs, 32, 16, 0, lane);
      o1 = __builtin_amdgcn_wmma_f32_16x16x32_bf16(false, pa, false, vb1,
                                                   (short)0, o1, false, false);
      bf16x16 vb2 = load_b_frag(Vs, 32, 32, 0, lane);
      o2 = __builtin_amdgcn_wmma_f32_16x16x32_bf16(false, pa, false, vb2,
                                                   (short)0, o2, false, false);
      bf16x16 vb3 = load_b_frag(Vs, 32, 48, 0, lane);
      o3 = __builtin_amdgcn_wmma_f32_16x16x32_bf16(false, pa, false, vb3,
                                                   (short)0, o3, false, false);
    }

    __syncthreads();                                 // before next overwrite
  }

  // ---- epilogue: 1/row_sum, query-mask, head merge, residual -----------
  int nloc = lane & 15;
#pragma unroll
  for (int r = 0; r < 8; ++r) {
    int   row = qbase + r + ((lane >> 4) << 3);
    float qm  = qmask[(size_t)b * SEQ + row];
    float inv = (row_sum[r] > 0.f) ? (qm / row_sum[r]) : 0.f;
    size_t base = ((size_t)b * SEQ + row) * DIN + (size_t)h * DEPTH;
    out[base + 0  + nloc] = o0[r] * inv + query[base + 0  + nloc];
    out[base + 16 + nloc] = o1[r] * inv + query[base + 16 + nloc];
    out[base + 32 + nloc] = o2[r] * inv + query[base + 32 + nloc];
    out[base + 48 + nloc] = o3[r] * inv + query[base + 48 + nloc];
  }
}

// ---------------------------------------------------------------------------
// Host launcher
// ---------------------------------------------------------------------------
extern "C" void kernel_launch(void* const* d_in, const int* in_sizes, int n_in,
                              void* d_out, int out_size, void* d_ws, size_t ws_size,
                              hipStream_t stream) {
  const float* query = (const float*)d_in[0];
  const float* key   = (const float*)d_in[1];
  const float* Wq    = (const float*)d_in[2];
  const float* Wk    = (const float*)d_in[3];
  const float* Wv    = (const float*)d_in[4];
  float* out = (float*)d_out;

  char* ws = (char*)d_ws;
  const size_t HBLD = (size_t)HEADS * BATCH * SEQ * DEPTH * sizeof(__bf16); // 16 MiB
  __bf16* Qh   = (__bf16*)(ws);
  __bf16* Kh   = (__bf16*)(ws + HBLD);
  __bf16* Vt   = (__bf16*)(ws + 2 * HBLD);
  __bf16* Wt   = (__bf16*)(ws + 3 * HBLD);                          // 1.5 MiB
  float*  qmsk = (float*)(ws + 3 * HBLD + 3ull * DIN * DIN * sizeof(__bf16));
  float*  kmsk = qmsk + (size_t)BATCH * SEQ;

  // Pass 0: masks + weight transpose/convert.
  mha_mask_kernel<<<dim3((BATCH * SEQ) / 8), dim3(256), 0, stream>>>(query, key, qmsk, kmsk);
  mha_wt_kernel<<<dim3((DIN * DIN) / 256, 3), dim3(256), 0, stream>>>(Wq, Wk, Wv, Wt);

  // Pass 1: Q/K/V projection GEMMs (16x64 tile per wave, 4 waves/block).
  mha_proj_kernel<<<dim3(256, DIN / 64, 3), dim3(128), 0, stream>>>(
      query, key, Wt, Qh, Kh, Vt);

  // Pass 2: streaming attention (grid: H*B x L/64, 4 waves/block).
  mha_attn_kernel<<<dim3(HEADS * BATCH, SEQ / 64), dim3(128), 0, stream>>>(
      Qh, Kh, Vt, qmsk, kmsk, query, out);
}
